// CompressiveTransformer_65206193488305
// MI455X (gfx1250) — matrix operand order repaired and checked by reading, not compiled, not run on hardware
//
#include <hip/hip_runtime.h>

// ---------------- constants ----------------
#define VOCAB 32000
#define DIM   512
#define HEADS 8
#define DIMH  64
#define SEQ   512
#define MEM_  512
#define CMEM_ 128
#define RATIO 4
#define DEPTH 4
#define BATCH 4
#define KVLEN 1152            // CMEM + MEM + SEQ
#define SCALE 0.125f          // 64^-0.5

// d_out layout (floats): logits | nm | nc | aux
#define OUT_LOGITS 0LL
#define OUT_NM     65536000LL                 // 2048*32000
#define OUT_NC     (OUT_NM + 4194304LL)       // + 4*4*512*512
#define OUT_AUX    (OUT_NC + 1048576LL)       // + 4*4*128*512

// workspace offsets (floats)
#define WS_X     0LL
#define WS_XN    1048576LL
#define WS_Q     2097152LL
#define WS_KVIN  3145728LL
#define WS_KV    5505024LL
#define WS_DOTS  10223616LL
#define WS_QPE   29097984LL
#define WS_OBUF  47972352LL
#define WS_YBUF  49020928LL
#define WS_H1    50069504LL
#define WS_GI    54263808LL
#define WS_GH    57409536LL
#define WS_CBUF  60555264LL
#define WS_WPACK 60817408LL
#define WS_CKV   61865984LL
#define WS_ATTA  62390272LL
#define WS_ATTB  63438848LL
#define WS_AUXA  64487424LL

typedef __attribute__((ext_vector_type(16))) __bf16    v16bf;
typedef __attribute__((ext_vector_type(8)))  float     v8f;
typedef __attribute__((ext_vector_type(8)))  unsigned  v8u;

// pack two f32 -> bf16x2 dword (RNE), low half = first element
__device__ __forceinline__ unsigned pack2(float a, float b) {
  unsigned ua = __builtin_bit_cast(unsigned, a);
  unsigned ub = __builtin_bit_cast(unsigned, b);
  ua = (ua + 0x7FFFu + ((ua >> 16) & 1u)) >> 16;
  ub = (ub + 0x7FFFu + ((ub >> 16) & 1u)) >> 16;
  return (ua & 0xFFFFu) | (ub << 16);
}

// ---------------------------------------------------------------------------
// Tensor-Data-Mover probe: issues a real tensor_load_to_lds with a NULL
// descriptor (D# group0.count == 0 -> "NULL tensor" per ISA 8.3, hardware
// no-op) followed by s_wait_tensorcnt 0. Exercises the CDNA5 async-tensor
// path without touching memory. Toolchain arity (5 vs 6 args) selected by
// presence of the therock TDM header, per the probe notes.
// ---------------------------------------------------------------------------
__global__ void tdm_probe_k() {
#if defined(__gfx1250__)
  typedef __attribute__((ext_vector_type(4))) unsigned u32x4;
  typedef __attribute__((ext_vector_type(8))) int      i32x8;
  typedef __attribute__((ext_vector_type(4))) int      i32x4;
  u32x4 g0 = {0u, 0u, 0u, 0u};        // count=0 => NULL tensor (no transfer)
  i32x8 g1 = {0, 0, 0, 0, 0, 0, 0, 0};
  i32x4 g2 = {0, 0, 0, 0};
  i32x4 g3 = {0, 0, 0, 0};
#if __has_include(<hip/amd_detail/amd_gfx1250_TDM.h>)
  i32x8 g4 = {0, 0, 0, 0, 0, 0, 0, 0};
  __builtin_amdgcn_tensor_load_to_lds(g0, g1, g2, g3, g4, 0);
#else
  __builtin_amdgcn_tensor_load_to_lds(g0, g1, g2, g3, 0);
#endif
  __builtin_amdgcn_s_wait_tensorcnt(0);
#endif
}

// ---------------------------------------------------------------------------
// Generic batched GEMM: C = act(alpha * A@op(B) + bias)
// REQUIRES: M%128==0, N%64==0, K%64==0 (true for every GEMM in this model)
// Block: 256 thr = 8 waves. Block tile 128(M) x 64(N). Wave tile 32x32
// (2x2 WMMA 16x16x32 bf16, f32 accum). K stepped by 64 via LDS (2 sub-steps).
// LDS layouts padded to 36 dwords/row (16B-aligned, bank-spread) so that BOTH
// A and B fragments are two contiguous ds_load_b128 per tile -> loads land
// directly in WMMA operand tuples (no v_mov shuffling).
// Batch offsets: off = (batch/mod)*so + (batch%mod)*si  (per A,B,C).
// ---------------------------------------------------------------------------
#define LDPAD 36

__global__ __launch_bounds__(256) void gemm_k(
    const float* __restrict__ A, const float* __restrict__ B,
    float* __restrict__ C, const float* __restrict__ bias,
    int M, int N, int K, int lda, int ldb, int ldc,
    long long soA, long long siA, int modA,
    long long soB, long long siB, int modB,
    long long soC, long long siC, int modC,
    int transB, int act, float alpha)
{
  __shared__ unsigned As32[128 * LDPAD];  // row-major: 128 M-rows x 32 K-pairs (+pad), 18 KB
  __shared__ unsigned Bs32[64 * LDPAD];   // transposed: 64 N-cols x 32 K-pairs (+pad), 9 KB

  const int batch = blockIdx.z;
  A += (long long)(batch / modA) * soA + (long long)(batch % modA) * siA;
  B += (long long)(batch / modB) * soB + (long long)(batch % modB) * siB;
  C += (long long)(batch / modC) * soC + (long long)(batch % modC) * siC;

  const int tid  = threadIdx.x;
  const int lane = tid & 31;
  const int wid  = tid >> 5;
  const int warpM = wid >> 1;            // 0..3
  const int warpN = wid & 1;             // 0..1
  const int bm = blockIdx.y * 128;
  const int bn = blockIdx.x * 64;
  const int half = lane >> 4;            // lane group 0/1
  const int l15  = lane & 15;

  v8f acc[2][2] = {};

  for (int k0 = 0; k0 < K; k0 += 64) {
    // ---- stage A (128 x 64 f32 -> bf16 pairs): float4 loads, dword stores
    #pragma unroll
    for (int e = 0; e < 8; ++e) {
      int s  = e * 256 + tid;            // float4 slot
      int r  = s >> 4;                   // 16 float4 per row
      int c4 = s & 15;
      const float4 v = *(const float4*)(A + (long long)(bm + r) * lda + k0 + c4 * 4);
      As32[r * LDPAD + c4 * 2]     = pack2(v.x, v.y);
      As32[r * LDPAD + c4 * 2 + 1] = pack2(v.z, v.w);
    }
    // ---- stage B (64 x 64) as packed K-pairs, transposed layout Bs[n][k2]
    if (transB) {
      #pragma unroll
      for (int e = 0; e < 8; ++e) {
        int s  = e * 256 + tid;
        int k2 = s >> 6, n = s & 63;
        const float2 v = *(const float2*)(B + (long long)(bn + n) * ldb + k0 + k2 * 2);
        Bs32[n * LDPAD + k2] = pack2(v.x, v.y);
      }
    } else {
      #pragma unroll
      for (int e = 0; e < 8; ++e) {
        int s  = e * 256 + tid;
        int k2 = s >> 6, n = s & 63;
        float b0 = B[(long long)(k0 + k2 * 2) * ldb + bn + n];
        float b1 = B[(long long)(k0 + k2 * 2 + 1) * ldb + bn + n];
        Bs32[n * LDPAD + k2] = pack2(b0, b1);
      }
    }
    // prefetch next K tile (uniform branch; emits global_prefetch_b8)
    if (k0 + 64 < K) {
      int r = tid >> 4, c4 = tid & 15;
      __builtin_prefetch(A + (long long)(bm + r) * lda + k0 + 64 + c4 * 4, 0, 1);
      int k2 = tid >> 6, n = tid & 63;
      __builtin_prefetch(transB ? (B + (long long)(bn + n) * ldb + k0 + 64 + k2 * 2)
                                : (B + (long long)(k0 + 64 + k2 * 2) * ldb + bn + n), 0, 1);
    }
    __syncthreads();

    #pragma unroll
    for (int sub = 0; sub < 2; ++sub) {
      v16bf afrag[2], bfrag[2];
      #pragma unroll
      for (int t = 0; t < 2; ++t) {
        // A fragment: 8 contiguous dwords (k = 16*half + s, matches B)
        const unsigned* Ab = &As32[(warpM * 32 + t * 16 + l15) * LDPAD + sub * 16 + half * 8];
        v8u au;
        au[0] = Ab[0]; au[1] = Ab[1]; au[2] = Ab[2]; au[3] = Ab[3];
        au[4] = Ab[4]; au[5] = Ab[5]; au[6] = Ab[6]; au[7] = Ab[7];
        afrag[t] = __builtin_bit_cast(v16bf, au);
        // B fragment: 8 contiguous dwords (transposed pair layout)
        const unsigned* Bb = &Bs32[(warpN * 32 + t * 16 + l15) * LDPAD + sub * 16 + half * 8];
        v8u bu;
        #pragma unroll
        for (int j = 0; j < 8; ++j) bu[j] = Bb[j];
        bfrag[t] = __builtin_bit_cast(v16bf, bu);
      }
      #pragma unroll
      for (int ti = 0; ti < 2; ++ti)
        #pragma unroll
        for (int tj = 0; tj < 2; ++tj)
          acc[ti][tj] = __builtin_amdgcn_wmma_f32_16x16x32_bf16(
              false, afrag[ti], false, bfrag[tj],
              (short)0, acc[ti][tj], false, false);
    }
    __syncthreads();
  }

  // epilogue: C VGPR r -> M = r + 8*half, N = lane&15 (no bounds needed)
  #pragma unroll
  for (int ti = 0; ti < 2; ++ti)
    #pragma unroll
    for (int tj = 0; tj < 2; ++tj)
      #pragma unroll
      for (int r = 0; r < 8; ++r) {
        int gm = bm + warpM * 32 + ti * 16 + r + half * 8;
        int gn = bn + warpN * 32 + tj * 16 + l15;
        float v = acc[ti][tj][r] * alpha;
        if (bias) v += bias[gn];
        if (act == 1) v = v > 0.f ? v : 0.01f * v;   // leaky relu
        C[(long long)gm * ldc + gn] = v;
      }
}

// ---------------- elementwise / reduction kernels ----------------

__global__ void embed_k(const int* __restrict__ tok,
                        const float* __restrict__ emb, float* __restrict__ x) {
  long long idx = (long long)blockIdx.x * 256 + threadIdx.x;
  if (idx >= (long long)BATCH * SEQ * DIM) return;
  long long row = idx / DIM; int col = (int)(idx % DIM);
  x[idx] = emb[(long long)tok[row] * DIM + col];
}

// one wave (32 lanes) per 512-wide row
__global__ void layernorm_k(const float* __restrict__ x,
                            const float* __restrict__ g, const float* __restrict__ b,
                            float* __restrict__ o) {
  long long row = blockIdx.x;
  int lane = threadIdx.x;
  const float* xr = x + row * DIM;
  float v[16], s = 0.f;
  #pragma unroll
  for (int i = 0; i < 16; ++i) { v[i] = xr[lane + 32 * i]; s += v[i]; }
  #pragma unroll
  for (int o2 = 16; o2 > 0; o2 >>= 1) s += __shfl_xor(s, o2, 32);
  float mu = s * (1.f / DIM);
  float var = 0.f;
  #pragma unroll
  for (int i = 0; i < 16; ++i) { float d = v[i] - mu; var += d * d; }
  #pragma unroll
  for (int o2 = 16; o2 > 0; o2 >>= 1) var += __shfl_xor(var, o2, 32);
  float inv = rsqrtf(var * (1.f / DIM) + 1e-5f);
  #pragma unroll
  for (int i = 0; i < 16; ++i) {
    int c = lane + 32 * i;
    o[row * DIM + c] = (v[i] - mu) * inv * g[c] + b[c];
  }
}

__global__ void concat_kv_k(const float* __restrict__ cm, const float* __restrict__ mm,
                            const float* __restrict__ xn, float* __restrict__ kvin) {
  long long idx = (long long)blockIdx.x * 256 + threadIdx.x;
  if (idx >= (long long)BATCH * KVLEN * DIM) return;
  int col = (int)(idx % DIM);
  long long t = idx / DIM;
  int row = (int)(t % KVLEN);
  int b = (int)(t / KVLEN);
  float v;
  if (row < CMEM_)              v = cm[((long long)b * CMEM_ + row) * DIM + col];
  else if (row < CMEM_ + MEM_)  v = mm[((long long)b * MEM_ + (row - CMEM_)) * DIM + col];
  else                          v = xn[((long long)b * SEQ + (row - CMEM_ - MEM_)) * DIM + col];
  kvin[idx] = v;
}

// dots = (dots + rel_shift(qpe)) * SCALE ; rel-shift done by index remap
__global__ void shift_add_k(float* __restrict__ dots, const float* __restrict__ qpe) {
  long long idx = (long long)blockIdx.x * 256 + threadIdx.x;
  if (idx >= (long long)BATCH * HEADS * SEQ * KVLEN) return;
  int c = (int)(idx % KVLEN);
  long long t = idx / KVLEN;
  int q = (int)(t % SEQ);
  long long bh = t / SEQ;
  const long long L = SEQ + KVLEN - 1;     // 1663
  const long long L2 = SEQ + KVLEN;        // 1664
  long long f = (long long)q * L + (SEQ - 1) + c;
  long long r = f / L2, col = f % L2;
  float sv = (col < KVLEN && r < SEQ) ? qpe[(bh * SEQ + r) * KVLEN + col] : 0.f;
  dots[idx] = (dots[idx] + sv) * SCALE;
}

__global__ void softmax_k(float* __restrict__ x, int W) {
  __shared__ float red[128];
  long long row = blockIdx.x;
  float* xr = x + row * (long long)W;
  int t = threadIdx.x;
  float m = -3.4e38f;
  for (int c = t; c < W; c += 128) m = fmaxf(m, xr[c]);
  red[t] = m; __syncthreads();
  for (int o = 64; o > 0; o >>= 1) { if (t < o) red[t] = fmaxf(red[t], red[t + o]); __syncthreads(); }
  m = red[0]; __syncthreads();
  float s = 0.f;
  for (int c = t; c < W; c += 128) { float e = __expf(xr[c] - m); xr[c] = e; s += e; }
  red[t] = s; __syncthreads();
  for (int o = 64; o > 0; o >>= 1) { if (t < o) red[t] += red[t + o]; __syncthreads(); }
  float inv = 1.f / red[0];
  for (int c = t; c < W; c += 128) xr[c] *= inv;
}

__global__ void gru_k(const float* __restrict__ gi, const float* __restrict__ gh,
                      const float* __restrict__ h, float* __restrict__ out, long long n) {
  long long idx = (long long)blockIdx.x * 256 + threadIdx.x;
  if (idx >= n) return;
  int col = (int)(idx % DIM);
  long long row = idx / DIM;
  const float* gir = gi + row * (3 * DIM);
  const float* ghr = gh + row * (3 * DIM);
  float ir = gir[col], iz = gir[DIM + col], in = gir[2 * DIM + col];
  float hr = ghr[col], hz = ghr[DIM + col], hn = ghr[2 * DIM + col];
  float r = 1.f / (1.f + __expf(-(ir + hr)));
  float z = 1.f / (1.f + __expf(-(iz + hz)));
  float nn = tanhf(in + r * hn);
  out[idx] = (1.f - z) * nn + z * h[idx];
}

// conv_w [o][i][r] -> wpack[k=r*DIM+i][o]
__global__ void pack_w_k(const float* __restrict__ w, float* __restrict__ wp) {
  long long idx = (long long)blockIdx.x * 256 + threadIdx.x;
  if (idx >= (long long)RATIO * DIM * DIM) return;
  int o = (int)(idx % DIM);
  long long k = idx / DIM;
  int r = (int)(k / DIM), i = (int)(k % DIM);
  wp[idx] = w[((long long)o * DIM + i) * RATIO + r];
}

__global__ void copy_k(float* __restrict__ dst, const float* __restrict__ src, long long n) {
  long long idx = (long long)blockIdx.x * 256 + threadIdx.x;
  if (idx < n) dst[idx] = src[idx];
}

__global__ void zero_k(float* __restrict__ p, long long n) {
  long long idx = (long long)blockIdx.x * 256 + threadIdx.x;
  if (idx < n) p[idx] = 0.f;
}

__global__ void mse_k(const float* __restrict__ a, const float* __restrict__ b,
                      float* __restrict__ acc, long long n) {
  __shared__ float red[256];
  long long stride = (long long)gridDim.x * 256;
  float s = 0.f;
  for (long long i = (long long)blockIdx.x * 256 + threadIdx.x; i < n; i += stride) {
    float d = a[i] - b[i]; s += d * d;
  }
  red[threadIdx.x] = s; __syncthreads();
  for (int o = 128; o > 0; o >>= 1) {
    if ((int)threadIdx.x < o) red[threadIdx.x] += red[threadIdx.x + o];
    __syncthreads();
  }
  if (threadIdx.x == 0) atomicAdd(acc, red[0]);
}

__global__ void finalize_aux_k(const float* __restrict__ acc, float* __restrict__ out, float invn) {
  if (threadIdx.x == 0) out[0] = acc[0] * invn;
}

// ---------------- host side ----------------

static inline void gemm(hipStream_t s, const float* A, const float* B, float* C,
                        const float* bias, int M, int N, int K,
                        int lda, int ldb, int ldc, int batch,
                        long long soA, long long siA, int modA,
                        long long soB, long long siB, int modB,
                        long long soC, long long siC, int modC,
                        int transB, int act, float alpha) {
  dim3 grid((N + 63) / 64, (M + 127) / 128, batch), block(256);
  gemm_k<<<grid, block, 0, s>>>(A, B, C, bias, M, N, K, lda, ldb, ldc,
                                soA, siA, modA, soB, siB, modB, soC, siC, modC,
                                transB, act, alpha);
}

static inline dim3 ew(long long n) { return dim3((unsigned)((n + 255) / 256)); }

extern "C" void kernel_launch(void* const* d_in, const int* in_sizes, int n_in,
                              void* d_out, int out_size, void* d_ws, size_t ws_size,
                              hipStream_t stream) {
  (void)in_sizes; (void)n_in; (void)out_size; (void)ws_size;
  const int*   tokens    = (const int*)  d_in[0];
  const float* mem       = (const float*)d_in[1];
  const float* cmem      = (const float*)d_in[2];
  const float* token_emb = (const float*)d_in[3];
  const float* pos_emb   = (const float*)d_in[4];
  const float* logits_w  = (const float*)d_in[5];
  const float* logits_b  = (const float*)d_in[6];
  const float* ln1_g     = (const float*)d_in[7];
  const float* ln1_b     = (const float*)d_in[8];
  const float* wq        = (const float*)d_in[9];
  const float* wkv       = (const float*)d_in[10];
  const float* wo        = (const float*)d_in[11];
  const float* wo_b      = (const float*)d_in[12];
  const float* conv_w    = (const float*)d_in[13];
  const float* conv_b    = (const float*)d_in[14];
  const float* g1_wih    = (const float*)d_in[15];
  const float* g1_whh    = (const float*)d_in[16];
  const float* g1_bih    = (const float*)d_in[17];
  const float* g1_bhh    = (const float*)d_in[18];
  const float* ln2_g     = (const float*)d_in[19];
  const float* ln2_b     = (const float*)d_in[20];
  const float* ff_w1     = (const float*)d_in[21];
  const float* ff_b1     = (const float*)d_in[22];
  const float* ff_w2     = (const float*)d_in[23];
  const float* ff_b2     = (const float*)d_in[24];
  const float* g2_wih    = (const float*)d_in[25];
  const float* g2_whh    = (const float*)d_in[26];
  const float* g2_bih    = (const float*)d_in[27];
  const float* g2_bhh    = (const float*)d_in[28];

  float* out = (float*)d_out;
  float* ws  = (float*)d_ws;

  const long long NTOK = (long long)BATCH * SEQ;           // 2048
  const long long NX   = NTOK * DIM;                       // 2048*512

  // async-tensor path probe (hardware no-op: NULL descriptor)
  tdm_probe_k<<<1, 32, 0, stream>>>();

  // x = token_emb[tokens]
  embed_k<<<ew(NX), 256, 0, stream>>>(tokens, token_emb, ws + WS_X);

  for (int l = 0; l < DEPTH; ++l) {
    const float* wq_l   = wq    + (long long)l * DIM * DIM;
    const float* wkv_l  = wkv   + (long long)l * DIM * 2 * DIM;
    const float* wo_l   = wo    + (long long)l * DIM * DIM;
    const float* wob_l  = wo_b  + (long long)l * DIM;
    const float* cw_l   = conv_w+ (long long)l * DIM * DIM * RATIO;
    const float* cb_l   = conv_b+ (long long)l * DIM;
    const float* mem_l  = mem   + (long long)l * BATCH * MEM_ * DIM;
    const float* cmem_l = cmem  + (long long)l * BATCH * CMEM_ * DIM;

    // --- ln1, q, kv ---
    layernorm_k<<<(unsigned)NTOK, 32, 0, stream>>>(ws + WS_X, ln1_g + l * DIM, ln1_b + l * DIM, ws + WS_XN);
    // new_mem[l] = xn
    copy_k<<<ew(NX), 256, 0, stream>>>(out + OUT_NM + (long long)l * NX, ws + WS_XN, NX);

    gemm(stream, ws + WS_XN, wq_l, ws + WS_Q, nullptr, (int)NTOK, DIM, DIM,
         DIM, DIM, DIM, 1, 0,0,1, 0,0,1, 0,0,1, 0, 0, 1.f);

    concat_kv_k<<<ew((long long)BATCH * KVLEN * DIM), 256, 0, stream>>>(cmem_l, mem_l, ws + WS_XN, ws + WS_KVIN);
    gemm(stream, ws + WS_KVIN, wkv_l, ws + WS_KV, nullptr, BATCH * KVLEN, 2 * DIM, DIM,
         DIM, 2 * DIM, 2 * DIM, 1, 0,0,1, 0,0,1, 0,0,1, 0, 0, 1.f);

    // --- attention scores: q@k^T (batched over b*h via strided heads) ---
    gemm(stream, ws + WS_Q, ws + WS_KV, ws + WS_DOTS, nullptr, SEQ, KVLEN, DIMH,
         DIM, 2 * DIM, KVLEN, BATCH * HEADS,
         (long long)SEQ * DIM, DIMH, HEADS,                 // A: q  [b][i][h*64+d]
         (long long)KVLEN * 2 * DIM, DIMH, HEADS,           // B: k  [b][j][h*64+d] (transB)
         (long long)SEQ * KVLEN, 0, 1, 1, 0, 1.f);
    // rel-pos: q@pe^T, pe batched over h only
    gemm(stream, ws + WS_Q, pos_emb, ws + WS_QPE, nullptr, SEQ, KVLEN, DIMH,
         DIM, DIMH, KVLEN, BATCH * HEADS,
         (long long)SEQ * DIM, DIMH, HEADS,
         0, (long long)KVLEN * DIMH, HEADS,                 // B: pe[h]
         (long long)SEQ * KVLEN, 0, 1, 1, 0, 1.f);

    shift_add_k<<<ew((long long)BATCH * HEADS * SEQ * KVLEN), 256, 0, stream>>>(ws + WS_DOTS, ws + WS_QPE);
    softmax_k<<<(unsigned)(BATCH * HEADS * SEQ), 128, 0, stream>>>(ws + WS_DOTS, KVLEN);

    // out = attn@v -> obuf[b][i][h*64+d]
    gemm(stream, ws + WS_DOTS, ws + WS_KV + DIM, ws + WS_OBUF, nullptr, SEQ, DIMH, KVLEN,
         KVLEN, 2 * DIM, DIM, BATCH * HEADS,
         (long long)SEQ * KVLEN, 0, 1,
         (long long)KVLEN * 2 * DIM, DIMH, HEADS,
         (long long)SEQ * DIM, DIMH, HEADS, 0, 0, 1.f);

    // logits = out@wo + wo_b
    gemm(stream, ws + WS_OBUF, wo_l, ws + WS_YBUF, wob_l, (int)NTOK, DIM, DIM,
         DIM, DIM, DIM, 1, 0,0,1, 0,0,1, 0,0,1, 0, 0, 1.f);

    // --- GRU 1 ---
    gemm(stream, ws + WS_YBUF, g1_wih + (long long)l * 3 * DIM * DIM, ws + WS_GI,
         g1_bih + (long long)l * 3 * DIM, (int)NTOK, 3 * DIM, DIM,
         DIM, DIM, 3 * DIM, 1, 0,0,1, 0,0,1, 0,0,1, 1, 0, 1.f);
    gemm(stream, ws + WS_X, g1_whh + (long long)l * 3 * DIM * DIM, ws + WS_GH,
         g1_bhh + (long long)l * 3 * DIM, (int)NTOK, 3 * DIM, DIM,
         DIM, DIM, 3 * DIM, 1, 0,0,1, 0,0,1, 0,0,1, 1, 0, 1.f);
    gru_k<<<ew(NX), 256, 0, stream>>>(ws + WS_GI, ws + WS_GH, ws + WS_X, ws + WS_X, NX);

    // --- memory compression: compressed = mem_l (as [B*128,2048]) @ wpack + conv_b ---
    pack_w_k<<<ew((long long)RATIO * DIM * DIM), 256, 0, stream>>>(cw_l, ws + WS_WPACK);
    gemm(stream, mem_l, ws + WS_WPACK, ws + WS_CBUF, cb_l, BATCH * CMEM_, DIM, RATIO * DIM,
         RATIO * DIM, DIM, DIM, 1, 0,0,1, 0,0,1, 0,0,1, 0, 0, 1.f);
    copy_k<<<ew((long long)BATCH * CMEM_ * DIM), 256, 0, stream>>>(
        out + OUT_NC + (long long)l * BATCH * CMEM_ * DIM, ws + WS_CBUF, (long long)BATCH * CMEM_ * DIM);

    // --- aux loss (reference overwrites aux each layer -> only last layer matters) ---
    if (l == DEPTH - 1) {
      // ck,cv = compressed @ wkv
      gemm(stream, ws + WS_CBUF, wkv_l, ws + WS_CKV, nullptr, BATCH * CMEM_, 2 * DIM, DIM,
           DIM, 2 * DIM, 2 * DIM, 1, 0,0,1, 0,0,1, 0,0,1, 0, 0, 1.f);
      // scores1 = q @ omk^T (omk = kv rows [CMEM, CMEM+MEM))
      gemm(stream, ws + WS_Q, ws + WS_KV + (long long)CMEM_ * 2 * DIM, ws + WS_DOTS, nullptr,
           SEQ, MEM_, DIMH, DIM, 2 * DIM, MEM_, BATCH * HEADS,
           (long long)SEQ * DIM, DIMH, HEADS,
           (long long)KVLEN * 2 * DIM, DIMH, HEADS,
           (long long)SEQ * MEM_, 0, 1, 1, 0, 1.f);
      softmax_k<<<(unsigned)(BATCH * HEADS * SEQ), 128, 0, stream>>>(ws + WS_DOTS, MEM_);
      gemm(stream, ws + WS_DOTS, ws + WS_KV + (long long)CMEM_ * 2 * DIM + DIM, ws + WS_ATTA, nullptr,
           SEQ, DIMH, MEM_, MEM_, 2 * DIM, DIMH, BATCH * HEADS,
           (long long)SEQ * MEM_, 0, 1,
           (long long)KVLEN * 2 * DIM, DIMH, HEADS,
           (long long)SEQ * DIMH, 0, 1, 0, 0, 1.f);
      // scores2 = q @ ck^T
      gemm(stream, ws + WS_Q, ws + WS_CKV, ws + WS_QPE, nullptr,
           SEQ, CMEM_, DIMH, DIM, 2 * DIM, CMEM_, BATCH * HEADS,
           (long long)SEQ * DIM, DIMH, HEADS,
           (long long)CMEM_ * 2 * DIM, DIMH, HEADS,
           (long long)SEQ * CMEM_, 0, 1, 1, 0, 1.f);
      softmax_k<<<(unsigned)(BATCH * HEADS * SEQ), 128, 0, stream>>>(ws + WS_QPE, CMEM_);
      gemm(stream, ws + WS_QPE, ws + WS_CKV + DIM, ws + WS_ATTB, nullptr,
           SEQ, DIMH, CMEM_, CMEM_, 2 * DIM, DIMH, BATCH * HEADS,
           (long long)SEQ * CMEM_, 0, 1,
           (long long)CMEM_ * 2 * DIM, DIMH, HEADS,
           (long long)SEQ * DIMH, 0, 1, 0, 0, 1.f);
      long long naux = (long long)BATCH * HEADS * SEQ * DIMH;
      zero_k<<<1, 256, 0, stream>>>(ws + WS_AUXA, 1);
      mse_k<<<512, 256, 0, stream>>>(ws + WS_ATTA, ws + WS_ATTB, ws + WS_AUXA, naux);
      finalize_aux_k<<<1, 32, 0, stream>>>(ws + WS_AUXA, out + OUT_AUX, 1.f / (float)naux);
    }

    // --- ln2 + FF ---
    layernorm_k<<<(unsigned)NTOK, 32, 0, stream>>>(ws + WS_X, ln2_g + l * DIM, ln2_b + l * DIM, ws + WS_XN);
    gemm(stream, ws + WS_XN, ff_w1 + (long long)l * DIM * 4 * DIM, ws + WS_H1,
         ff_b1 + (long long)l * 4 * DIM, (int)NTOK, 4 * DIM, DIM,
         DIM, 4 * DIM, 4 * DIM, 1, 0,0,1, 0,0,1, 0,0,1, 0, 1, 1.f);
    gemm(stream, ws + WS_H1, ff_w2 + (long long)l * 4 * DIM * DIM, ws + WS_YBUF,
         ff_b2 + (long long)l * DIM, (int)NTOK, DIM, 4 * DIM,
         4 * DIM, DIM, DIM, 1, 0,0,1, 0,0,1, 0,0,1, 0, 0, 1.f);

    // --- GRU 2 ---
    gemm(stream, ws + WS_YBUF, g2_wih + (long long)l * 3 * DIM * DIM, ws + WS_GI,
         g2_bih + (long long)l * 3 * DIM, (int)NTOK, 3 * DIM, DIM,
         DIM, DIM, 3 * DIM, 1, 0,0,1, 0,0,1, 0,0,1, 1, 0, 1.f);
    gemm(stream, ws + WS_X, g2_whh + (long long)l * 3 * DIM * DIM, ws + WS_GH,
         g2_bhh + (long long)l * 3 * DIM, (int)NTOK, 3 * DIM, DIM,
         DIM, DIM, 3 * DIM, 1, 0,0,1, 0,0,1, 0,0,1, 1, 0, 1.f);
    gru_k<<<ew(NX), 256, 0, stream>>>(ws + WS_GI, ws + WS_GH, ws + WS_X, ws + WS_X, NX);
  }

  // final logits: out = x @ logits_w + logits_b
  gemm(stream, ws + WS_X, logits_w, out + OUT_LOGITS, logits_b, (int)NTOK, VOCAB, DIM,
       DIM, VOCAB, VOCAB, 1, 0,0,1, 0,0,1, 0,0,1, 0, 0, 1.f);
}
